// SwinTransformer3D_7249904796470
// MI455X (gfx1250) — matrix operand order repaired and checked by reading, not compiled, not run on hardware
//
#include <hip/hip_runtime.h>
#include <hip/hip_bf16.h>
#include <math.h>

// ---------------------------------------------------------------------------
// CDNA5 (gfx1250) wave32 WMMA building blocks
// ---------------------------------------------------------------------------
typedef __attribute__((ext_vector_type(16))) __bf16 v16bf;
typedef __attribute__((ext_vector_type(8)))  float  v8f;

// A-fragment (16x32 bf16, MxK): lanes 0-15 row M=lane, K = {0..7,16..23};
// lanes 16-31 row M=lane-16, K = {8..15,24..31}.   (ISA 7.12.2)
template <typename F>
__device__ inline v16bf frag_a(F f) {
  v16bf v;
  int half = (threadIdx.x & 31) >> 4;
#pragma unroll
  for (int j = 0; j < 8; ++j) {
    v[j]     = f(half * 8 + j);
    v[j + 8] = f(16 + half * 8 + j);
  }
  return v;
}

// B-fragment (32x16 bf16, KxN): lanes 0-15 col N=lane, K = {0..7,16..23};
// lanes 16-31 col N=lane-16, K = {8..15,24..31}.   (ISA 7.12.2 / 7.12.4)
template <typename F>
__device__ inline v16bf frag_b(F f) {
  v16bf v;
  int half = (threadIdx.x & 31) >> 4;
#pragma unroll
  for (int j = 0; j < 8; ++j) {
    v[j]     = f(half * 8 + j);
    v[j + 8] = f(16 + half * 8 + j);
  }
  return v;
}

__device__ inline float gelu_exact(float v) {
  return 0.5f * v * (1.f + erff(v * 0.70710678118654752f));
}
__device__ inline float sigmoidf_(float v) { return 1.f / (1.f + __expf(-v)); }

// low 32 bits of a flat pointer into LDS == LDS byte offset (ISA 10.2 aperture)
__device__ inline unsigned lds_off(const void* p) {
  return (unsigned)(size_t)p;
}
__device__ inline void async_load_b32(unsigned lds, const float* gptr) {
  asm volatile("global_load_async_to_lds_b32 %0, %1, off"
               :: "v"(lds), "v"((unsigned long long)(size_t)gptr)
               : "memory");
}
__device__ inline void wait_async0() {
  asm volatile("s_wait_asynccnt 0" ::: "memory");
}

// ---------------------------------------------------------------------------
// Tiled WMMA GEMM: C[M,N] = act(A[M,K] @ B + bias)
// A row-major f32.  B element (k,n) at B[k*ldbk + n*ldbn] (f32).
// M,N mult of 64; K mult of 32.  256 threads = 8 waves per 64x64 macro-tile;
// each wave owns one 16-row band and two 16x16 tiles (A fragment reused).
// A/B k-slices staged to LDS with GLOBAL_LOAD_ASYNC_TO_LDS_B32 (ASYNCcnt).
// act: 0 none, 1 gelu, 2 relu
// ---------------------------------------------------------------------------
__global__ __launch_bounds__(256) void gemm_wmma_kernel(
    const float* __restrict__ A, const float* __restrict__ B,
    const float* __restrict__ bias, float* __restrict__ C,
    int M, int N, int K, long ldbk, long ldbn, int act) {
  __shared__ float Ash[64][33];   // 64 rows x 32 k (+pad)
  __shared__ float Bsh[32][65];   // 32 k x 64 cols (+pad)
  int tid = threadIdx.x;
  int wave = tid >> 5;
  int lane = tid & 31;
  int r = lane & 15;
  long bn = (long)blockIdx.x * 64;
  long bm = (long)blockIdx.y * 64;
  int wtm = (wave >> 1) * 16;        // wave's row band in macro-tile
  int wtn0 = (wave & 1) * 32;        // wave's two col tiles: wtn0, wtn0+16
  v8f acc0 = {}, acc1 = {};

  for (int k0 = 0; k0 < K; k0 += 32) {
    // ---- stage A(64x32) and B(32x64) slices into LDS (async, coalesced) ----
#pragma unroll
    for (int e = 0; e < 8; ++e) {
      int l = e * 256 + tid;
      int ra = l >> 5, ka = l & 31;                        // A element
      async_load_b32(lds_off(&Ash[ra][ka]),
                     A + (bm + ra) * (long)K + (k0 + ka));
      int kb = l >> 6, nb = l & 63;                        // B element
      async_load_b32(lds_off(&Bsh[kb][nb]),
                     B + (long)(k0 + kb) * ldbk + (bn + nb) * ldbn);
    }
    wait_async0();
    __syncthreads();

    // ---- fragments from LDS (f32 -> bf16), two WMMAs sharing A ----
    v16bf af = frag_a([&](int kk) { return (__bf16)Ash[wtm + r][kk]; });
    v16bf b0 = frag_b([&](int kk) { return (__bf16)Bsh[kk][wtn0 + r]; });
    v16bf b1 = frag_b([&](int kk) { return (__bf16)Bsh[kk][wtn0 + 16 + r]; });
    acc0 = __builtin_amdgcn_wmma_f32_16x16x32_bf16(false, af, false, b0,
                                                   (short)0, acc0, false, false);
    acc1 = __builtin_amdgcn_wmma_f32_16x16x32_bf16(false, af, false, b1,
                                                   (short)0, acc1, false, false);
    __syncthreads();   // protect LDS before next stage
  }

  int half = lane >> 4;
  long n0 = bn + wtn0 + r;
  long n1 = n0 + 16;
  float bv0 = bias ? bias[n0] : 0.f;
  float bv1 = bias ? bias[n1] : 0.f;
#pragma unroll
  for (int i = 0; i < 8; ++i) {
    long m = bm + wtm + half * 8 + i;
    float v0 = acc0[i] + bv0;
    float v1 = acc1[i] + bv1;
    if (act == 1) { v0 = gelu_exact(v0); v1 = gelu_exact(v1); }
    else if (act == 2) { v0 = fmaxf(v0, 0.f); v1 = fmaxf(v1, 0.f); }
    C[m * (long)N + n0] = v0;
    C[m * (long)N + n1] = v1;
  }
}

// ---------------------------------------------------------------------------
// Fused window attention: one block per (window, head). 64 tokens/window.
// qkv: (nWin*64, 3C) row-major (q|k|v).  out: (nWin*64, C), this head's cols.
// ---------------------------------------------------------------------------
__global__ __launch_bounds__(256) void attn_kernel(
    const float* __restrict__ qkv, const float* __restrict__ rpb,
    const int* __restrict__ rel, float* __restrict__ out,
    int C, int hd, float scale) {
  int win = blockIdx.x, h = blockIdx.y;
  int wave = threadIdx.x >> 5;
  int lane = threadIdx.x & 31;
  int r = lane & 15;
  long ld = 3L * C;
  const float* Q = qkv + (long)win * 64 * ld + (long)h * hd;
  const float* Kp = Q + C;
  const float* V = Q + 2 * C;
  __shared__ float S[64][65];

  // S = scale*Q @ K^T + bias      (4x4 tiles of 16x16, K = hd padded to 32)
  for (int t = wave; t < 16; t += 8) {
    int tm = (t >> 2) * 16, tn = (t & 3) * 16;
    v8f acc = {};
    for (int k0 = 0; k0 < hd; k0 += 32) {
      v16bf a = frag_a([&](int kk) {
        int d = k0 + kk;
        return (__bf16)(d < hd ? Q[(long)(tm + r) * ld + d] * scale : 0.f);
      });
      v16bf b = frag_b([&](int kk) {
        int d = k0 + kk;
        return (__bf16)(d < hd ? Kp[(long)(tn + r) * ld + d] : 0.f);
      });
      acc = __builtin_amdgcn_wmma_f32_16x16x32_bf16(false, a, false, b, (short)0,
                                                    acc, false, false);
    }
    int half = lane >> 4;
    int n = tn + r;
#pragma unroll
    for (int i = 0; i < 8; ++i) {
      int m = tm + half * 8 + i;
      S[m][n] = acc[i] + rpb[rel[m * 64 + n] * 8 + h];
    }
  }
  __syncthreads();

  // softmax per row
  if (threadIdx.x < 64) {
    int m = threadIdx.x;
    float mx = -1e30f;
    for (int j = 0; j < 64; ++j) mx = fmaxf(mx, S[m][j]);
    float s = 0.f;
    for (int j = 0; j < 64; ++j) { float e = __expf(S[m][j] - mx); S[m][j] = e; s += e; }
    float inv = 1.f / s;
    for (int j = 0; j < 64; ++j) S[m][j] *= inv;
  }
  __syncthreads();

  // O = P @ V   (4 x hd/16 tiles, K = 64)
  int ntn = hd / 16;
  for (int t = wave; t < 4 * ntn; t += 8) {
    int tm = (t / ntn) * 16, tn = (t % ntn) * 16;
    v8f acc = {};
    for (int k0 = 0; k0 < 64; k0 += 32) {
      v16bf a = frag_a([&](int kk) { return (__bf16)S[tm + r][k0 + kk]; });
      v16bf b = frag_b([&](int kk) {
        return (__bf16)V[(long)(k0 + kk) * ld + tn + r];
      });
      acc = __builtin_amdgcn_wmma_f32_16x16x32_bf16(false, a, false, b, (short)0,
                                                    acc, false, false);
    }
    int half = lane >> 4;
    int n = tn + r;
#pragma unroll
    for (int i = 0; i < 8; ++i) {
      int m = tm + half * 8 + i;
      out[((long)win * 64 + m) * (long)C + h * hd + n] = acc[i];
    }
  }
}

// ---------------------------------------------------------------------------
// LayerNorm with optional residual: y = LN(x + res) * g + b   (row length C)
// ---------------------------------------------------------------------------
__global__ __launch_bounds__(256) void ln_kernel(
    const float* __restrict__ x, const float* __restrict__ res,
    const float* __restrict__ g, const float* __restrict__ b,
    float* __restrict__ y, int C) {
  long row = blockIdx.x;
  const float* xr = x + row * (long)C;
  const float* rr = res ? res + row * (long)C : nullptr;
  float* yr = y + row * (long)C;
  float s = 0.f, s2 = 0.f;
  for (int j = threadIdx.x; j < C; j += 256) {
    float v = xr[j] + (rr ? rr[j] : 0.f);
    s += v; s2 += v * v;
  }
  __shared__ float sh1[256], sh2[256];
  sh1[threadIdx.x] = s; sh2[threadIdx.x] = s2;
  __syncthreads();
  for (int off = 128; off > 0; off >>= 1) {
    if (threadIdx.x < off) {
      sh1[threadIdx.x] += sh1[threadIdx.x + off];
      sh2[threadIdx.x] += sh2[threadIdx.x + off];
    }
    __syncthreads();
  }
  float mean = sh1[0] / C;
  float var = sh2[0] / C - mean * mean;
  float inv = rsqrtf(var + 1e-5f);
  for (int j = threadIdx.x; j < C; j += 256) {
    float v = xr[j] + (rr ? rr[j] : 0.f);
    yr[j] = (v - mean) * inv * g[j] + b[j];
  }
}

// ---------------------------------------------------------------------------
// Layout / permutation kernels (cubic spatial grids, B=1)
// ---------------------------------------------------------------------------
__global__ void win_partition_kernel(const float* __restrict__ x,
                                     float* __restrict__ t, int C, int D,
                                     int shift) {
  long i = (long)blockIdx.x * 256 + threadIdx.x;
  long total = (long)C * D * D * D;
  if (i >= total) return;
  int c = (int)(i % C);
  long tok = i / C;
  int sw = tok & 3, sh = (tok >> 2) & 3, sd = (tok >> 4) & 3;
  long win = tok >> 6;
  int nw = D >> 2;
  int wwi = (int)(win % nw); win /= nw;
  int whi = (int)(win % nw);
  int wdi = (int)(win / nw);
  int d = (wdi * 4 + sd - shift + D) % D;
  int h = (whi * 4 + sh - shift + D) % D;
  int w = (wwi * 4 + sw - shift + D) % D;
  t[i] = x[(((long)c * D + d) * D + h) * D + w];
}

__global__ void win_reverse_kernel(const float* __restrict__ t,
                                   float* __restrict__ x, int C, int D,
                                   int shift) {
  long i = (long)blockIdx.x * 256 + threadIdx.x;
  long total = (long)C * D * D * D;
  if (i >= total) return;
  int c = (int)(i % C);
  long tok = i / C;
  int sw = tok & 3, sh = (tok >> 2) & 3, sd = (tok >> 4) & 3;
  long win = tok >> 6;
  int nw = D >> 2;
  int wwi = (int)(win % nw); win /= nw;
  int whi = (int)(win % nw);
  int wdi = (int)(win / nw);
  int d = (wdi * 4 + sd - shift + D) % D;
  int h = (whi * 4 + sh - shift + D) % D;
  int w = (wwi * 4 + sw - shift + D) % D;
  x[(((long)c * D + d) * D + h) * D + w] = t[i];
}

// spatial (C,P) -> tokens (P,C)
__global__ void spatial_to_tok(const float* __restrict__ x,
                               float* __restrict__ t, int C, long P) {
  long i = (long)blockIdx.x * 256 + threadIdx.x;
  if (i >= (long)C * P) return;
  long c = i % C, p = i / C;
  t[i] = x[c * P + p];
}

// tokens (P,C) -> spatial (C,P)
__global__ void tok_to_spatial(const float* __restrict__ t,
                               float* __restrict__ x, int C, long P) {
  long i = (long)blockIdx.x * 256 + threadIdx.x;
  if (i >= (long)C * P) return;
  long c = i / P, p = i % P;
  x[i] = t[p * (long)C + c];
}

// patch-merge gather: spatial (C,D,D,D) -> rows ((D/2)^3, 8C)
__global__ void merge_gather_kernel(const float* __restrict__ x,
                                    float* __restrict__ out, int C, int D) {
  int D2 = D >> 1;
  long M = (long)D2 * D2 * D2;
  long K8 = 8L * C;
  long i = (long)blockIdx.x * 256 + threadIdx.x;
  if (i >= M * K8) return;
  long col = i % K8, row = i / K8;
  int c = (int)(col % C);
  int s = (int)(col / C);
  int sw = s & 1, sh = (s >> 1) & 1, sd = s >> 2;
  int w2 = (int)(row % D2);
  int h2 = (int)((row / D2) % D2);
  int d2 = (int)(row / ((long)D2 * D2));
  out[i] = x[(((long)c * D + (2 * d2 + sd)) * D + (2 * h2 + sh)) * D + (2 * w2 + sw)];
}

// ---------------------------------------------------------------------------
// Small kernels: rel-index table, patch-embed conv, im2col, heads
// ---------------------------------------------------------------------------
__global__ void relidx_kernel(int* rel) {
  int i = blockIdx.x * 256 + threadIdx.x;
  if (i >= 4096) return;
  int q = i >> 6, k = i & 63;
  int qd = q >> 4, qh = (q >> 2) & 3, qw = q & 3;
  int kd = k >> 4, kh = (k >> 2) & 3, kw = k & 3;
  rel[i] = (qd - kd + 3) * 49 + (qh - kh + 3) * 7 + (qw - kw + 3);
}

__global__ void patch_embed_kernel(const float* __restrict__ xin,
                                   const float* __restrict__ w,
                                   const float* __restrict__ b,
                                   float* __restrict__ X) {
  long i = (long)blockIdx.x * 256 + threadIdx.x;
  if (i >= 128L * 32768) return;
  int co = (int)(i / 32768);
  int p = (int)(i % 32768);
  int ow = p & 31, oh = (p >> 5) & 31, od = p >> 10;
  const float* wc = w + co * 64;
  float s = 0.f;
#pragma unroll
  for (int kd = 0; kd < 4; ++kd)
#pragma unroll
    for (int kh = 0; kh < 4; ++kh)
#pragma unroll
      for (int kw = 0; kw < 4; ++kw)
        s += xin[((long)(od * 4 + kd) * 128 + (oh * 4 + kh)) * 128 + (ow * 4 + kw)] *
             wc[kd * 16 + kh * 4 + kw];
  X[i] = s + b[co];
}

// im2col for 3x3x3 pad-1 conv on (1024,4,4,4) -> (64, 27648)
__global__ void im2col_kernel(const float* __restrict__ X, float* __restrict__ col) {
  long i = (long)blockIdx.x * 256 + threadIdx.x;
  if (i >= 64L * 27648) return;
  int k = (int)(i % 27648);
  int p = (int)(i / 27648);
  int ci = k / 27, rmd = k % 27;
  int kd = rmd / 9, kh = (rmd / 3) % 3, kw = rmd % 3;
  int w = p & 3, h = (p >> 2) & 3, d = p >> 4;
  int dd = d + kd - 1, hh = h + kh - 1, ww = w + kw - 1;
  float v = 0.f;
  if (dd >= 0 && dd < 4 && hh >= 0 && hh < 4 && ww >= 0 && ww < 4)
    v = X[((ci * 4 + dd) * 4 + hh) * 4 + ww];
  col[i] = v;
}

__global__ void seg_kernel(const float* __restrict__ X, const float* __restrict__ w,
                           const float* __restrict__ b, float* __restrict__ out,
                           int C, int P) {
  int p = threadIdx.x;
  if (p >= P) return;
  float s = 0.f;
  for (int c = 0; c < C; ++c) s += X[(long)c * P + p] * w[c];
  out[p] = sigmoidf_(s + b[0]);
}

__global__ void unc2_kernel(const float* __restrict__ U, const float* __restrict__ w,
                            const float* __restrict__ b, float* __restrict__ out,
                            int C, int P) {
  int p = threadIdx.x;
  if (p >= P) return;
  float s = 0.f;
  for (int c = 0; c < C; ++c) s += U[(long)p * C + c] * w[c];
  out[p] = sigmoidf_(s + b[0]);
}

// ---------------------------------------------------------------------------
// Host orchestration
// ---------------------------------------------------------------------------
extern "C" void kernel_launch(void* const* d_in, const int* in_sizes, int n_in,
                              void* d_out, int out_size, void* d_ws, size_t ws_size,
                              hipStream_t stream) {
  (void)in_sizes; (void)n_in; (void)out_size; (void)ws_size;

  int ii = 0;
  auto F = [&]() { return (const float*)d_in[ii++]; };

  struct Attn { const float *qw, *qb, *pw, *pb, *rpb; };
  struct Blk  { Attn a; const float *ang, *anb, *w1, *b1, *w2, *b2, *mng, *mnb;
                bool sh; Attn sa; const float *sg, *sb; };
  struct Stage { Blk b0, b1; const float *mg, *mb, *mw; };

  auto readAttn = [&]() { Attn a; a.qw = F(); a.qb = F(); a.pw = F(); a.pb = F(); a.rpb = F(); return a; };
  auto readBlk = [&](bool sh) {
    Blk b; b.a = readAttn();
    b.ang = F(); b.anb = F();
    b.w1 = F(); b.b1 = F(); b.w2 = F(); b.b2 = F();
    b.mng = F(); b.mnb = F();
    b.sh = sh;
    if (sh) { b.sa = readAttn(); b.sg = F(); b.sb = F(); }
    return b;
  };

  const float* x_in = F();
  const float* pe_w = F();
  const float* pe_b = F();
  Stage st[4];
  for (int s = 0; s < 4; ++s) {
    st[s].b0 = readBlk(false);
    st[s].b1 = readBlk(true);
    if (s < 3) { st[s].mg = F(); st[s].mb = F(); st[s].mw = F(); }
  }
  const float* seg_w = F(); const float* seg_b = F();
  const float* uw1 = F(); const float* ub1 = F();
  const float* uw2 = F(); const float* ub2 = F();

  // workspace layout
  const size_t MB = 1ull << 20;
  char* ws = (char*)d_ws;
  int*   rel = (int*)ws;                      // 16 KB
  float* X   = (float*)(ws + 1 * MB);         // 16 MB  spatial (C,P)
  float* T0  = (float*)(ws + 18 * MB);        // 16 MB  tokens
  float* AO  = (float*)(ws + 35 * MB);        // 16 MB  attn out / LN out
  float* PR  = (float*)(ws + 52 * MB);        // 16 MB  proj / mlp2 out
  float* BIG = (float*)(ws + 69 * MB);        // 68 MB  qkv / mlp hidden / gathers

  auto eb = [](long n) { return dim3((unsigned)((n + 255) / 256)); };
  auto gemm = [&](const float* A, const float* B, const float* bias, float* Cp,
                  int M, int N, int K, long ldbk, long ldbn, int act) {
    gemm_wmma_kernel<<<dim3((unsigned)(N / 64), (unsigned)(M / 64)), dim3(256), 0,
                       stream>>>(A, B, bias, Cp, M, N, K, ldbk, ldbn, act);
  };

  relidx_kernel<<<16, 256, 0, stream>>>(rel);
  patch_embed_kernel<<<eb(128L * 32768), 256, 0, stream>>>(x_in, pe_w, pe_b, X);

  auto attnPass = [&](int C, int D, const Attn& ap, const float* gg,
                      const float* bb, int shift) {
    long tok = (long)D * D * D;
    long tot = tok * C;
    int hd = C / 8;
    win_partition_kernel<<<eb(tot), 256, 0, stream>>>(X, T0, C, D, shift);
    gemm(T0, ap.qw, ap.qb, BIG, (int)tok, 3 * C, C, 3L * C, 1, 0);
    attn_kernel<<<dim3((unsigned)(tok / 64), 8), 256, 0, stream>>>(
        BIG, ap.rpb, rel, AO, C, hd, 1.0f / sqrtf((float)hd));
    gemm(AO, ap.pw, ap.pb, PR, (int)tok, C, C, (long)C, 1, 0);
    ln_kernel<<<dim3((unsigned)tok), 256, 0, stream>>>(T0, PR, gg, bb, AO, C);
    win_reverse_kernel<<<eb(tot), 256, 0, stream>>>(AO, X, C, D, shift);
  };

  auto runBlock = [&](const Blk& bk, int C, int D) {
    attnPass(C, D, bk.a, bk.ang, bk.anb, 0);
    if (bk.sh) attnPass(C, D, bk.sa, bk.sg, bk.sb, 2);
    long tok = (long)D * D * D;
    long tot = tok * C;
    spatial_to_tok<<<eb(tot), 256, 0, stream>>>(X, T0, C, tok);
    gemm(T0, bk.w1, bk.b1, BIG, (int)tok, 4 * C, C, 4L * C, 1, 1);       // GELU
    gemm(BIG, bk.w2, bk.b2, PR, (int)tok, C, 4 * C, (long)C, 1, 0);
    ln_kernel<<<dim3((unsigned)tok), 256, 0, stream>>>(T0, PR, bk.mng, bk.mnb, AO, C);
    tok_to_spatial<<<eb(tot), 256, 0, stream>>>(AO, X, C, tok);
  };

  auto runMerge = [&](const Stage& s, int C, int D) {
    int D2 = D >> 1;
    long M = (long)D2 * D2 * D2;
    long K8 = 8L * C;
    merge_gather_kernel<<<eb(M * K8), 256, 0, stream>>>(X, BIG, C, D);
    ln_kernel<<<dim3((unsigned)M), 256, 0, stream>>>(BIG, nullptr, s.mg, s.mb, AO,
                                                     (int)K8);
    gemm(AO, s.mw, nullptr, T0, (int)M, 2 * C, (int)K8, 2L * C, 1, 0);
    tok_to_spatial<<<eb(M * 2 * C), 256, 0, stream>>>(T0, X, 2 * C, M);
  };

  int C = 128, D = 32;
  for (int s = 0; s < 4; ++s) {
    runBlock(st[s].b0, C, D);
    runBlock(st[s].b1, C, D);
    if (s < 3) { runMerge(st[s], C, D); C *= 2; D /= 2; }
  }

  // heads: X is (1024, 4,4,4)
  float* outp = (float*)d_out;
  seg_kernel<<<1, 64, 0, stream>>>(X, seg_w, seg_b, outp, 1024, 64);
  im2col_kernel<<<eb(64L * 27648), 256, 0, stream>>>(X, BIG);
  gemm(BIG, uw1, ub1, T0, 64, 512, 27648, 1, 27648, 2);                  // ReLU
  unc2_kernel<<<1, 64, 0, stream>>>(T0, uw2, ub2, outp + 64, 512, 64);
}
